// AutoregressiveFlow_63591285784759
// MI455X (gfx1250) — matrix operand order repaired and checked by reading, not compiled, not run on hardware
//
#include <hip/hip_runtime.h>
#include <hip/hip_bf16.h>
#include <hip/hip_fp16.h>

typedef __attribute__((ext_vector_type(16))) _Float16 v16h;
typedef __attribute__((ext_vector_type(8)))  float    v8f;

#define BTOT   65536
#define LSEQ   64
#define NLAY   63          // L-1 steps
#define HD     64
#define MMIX   6
#define WAVES  4
#define NTHR   (WAVES * 32)
#define ROWB   144         // 72 halves per padded row (bytes)

static constexpr float F_PI     = 3.14159265358979323846f;
static constexpr float F_PI_2   = 1.57079632679489661923f;
static constexpr float F_2PI    = 6.28318530717958647692f;
static constexpr float F_INV2PI = 0.15915494309189533577f;

// ---------------- LDS layout (byte offsets into dynamic smem) ----------------
#define OFF_WT2   0                         // 64 rows * 144B  = 9216
#define OFF_WT3   9216                      // 32 rows * 144B  = 4608
#define OFF_W1    13824                     // 128 floats      = 512
#define OFF_B1    14336                     // 64 floats       = 256
#define OFF_B2    14592                     // 64 floats       = 256
#define OFF_B3    14848                     // 32 floats       = 128
#define OFF_H1    14976                     // 4*32*144        = 18432
#define OFF_H2    33408                     // 4*32*144        = 18432
#define OFF_OROW  51840                     // 4*32*20*4       = 10240
#define SMEM_SZ   62080

__device__ __forceinline__ float mod2pi(float x) {
    return x - F_2PI * floorf(x * F_INV2PI);
}
// tanh(x) = 1 - 2/(exp(2x)+1): v_mul, v_exp, v_add, v_rcp, v_fma (no IEEE div)
__device__ __forceinline__ float fast_tanh(float x) {
    float xx = fminf(fmaxf(x, -15.f), 15.f);        // v_med3
    float e  = __expf(2.f * xx);
    float r  = __builtin_amdgcn_rcpf(e + 1.f);
    return fmaf(-2.f, r, 1.f);
}
__device__ __forceinline__ float sigm2(float t) {   // sigmoid(2t), RAMP_POW = 2
    return __builtin_amdgcn_rcpf(1.f + __expf(-2.f * t));
}
__device__ __forceinline__ float softplus_f(float x) {
    return (x > 15.f) ? x : __logf(1.f + __expf(x));
}
// polynomial atan2 (v_rcp + 4 FMA + fixups), good to ~1e-5 rad
__device__ __forceinline__ float fast_atan2(float y, float x) {
    float ax = fabsf(x), ay = fabsf(y);
    float mx = fmaxf(ax, ay), mn = fminf(ax, ay);
    float a  = mn * __builtin_amdgcn_rcpf(mx);
    float s  = a * a;
    float r  = fmaf(s, fmaf(s, fmaf(s, fmaf(s, 0.0208351f, -0.0851330f),
                                    0.1801410f), -0.3302995f), 0.9998660f);
    r = r * a;
    if (ay > ax)  r = F_PI_2 - r;
    if (x < 0.f)  r = F_PI - r;
    return copysignf(r, y);
}

// A fragment (16x32 f16): lane = (M = ln&15, hi = ln>>4);
// elements e=0..7  -> K = kc*32 + 8*hi + e             (contiguous 16B)
// elements e=8..15 -> K = kc*32 + 16 + 8*hi + (e-8)    (contiguous 16B)
__device__ __forceinline__ v16h loadA(const char* base, int row, int kc, int hi) {
    union { v16h v; uint4 q[2]; } f;
    const char* r = base + row * ROWB;
    f.q[0] = *(const uint4*)(r + kc * 64 + hi * 16);
    f.q[1] = *(const uint4*)(r + kc * 64 + 32 + hi * 16);
    return f.v;
}
// B fragment (32x16 f16): lane = (N = ln&15, hi = ln>>4); same K order as A:
// e<8 -> K = kc*32 + 8*hi + e, e>=8 -> K = kc*32 + 16 + 8*hi + (e-8)
__device__ __forceinline__ v16h loadB(const char* base, int ncol, int kc, int hi) {
    union { v16h v; uint4 q[2]; } f;
    const char* r = base + ncol * ROWB;
    f.q[0] = *(const uint4*)(r + kc * 64 + hi * 16);
    f.q[1] = *(const uint4*)(r + kc * 64 + 32 + hi * 16);
    return f.v;
}
__device__ __forceinline__ v8f wmma_f16(v16h a, v16h b, v8f c) {
    return __builtin_amdgcn_wmma_f32_16x16x32_f16(false, a, false, b, (short)0, c,
                                                  false, false);
}

__global__ __launch_bounds__(NTHR, 1)
void flow_kernel(const float* __restrict__ z,
                 const float* __restrict__ W1, const float* __restrict__ b1,
                 const float* __restrict__ W2, const float* __restrict__ b2,
                 const float* __restrict__ W3, const float* __restrict__ b3,
                 float* __restrict__ out_phi, float* __restrict__ out_ldj) {
    extern __shared__ char smem[];
    char*  wt2b  = smem + OFF_WT2;
    char*  wt3b  = smem + OFF_WT3;
    float* W1s   = (float*)(smem + OFF_W1);
    float* b1s   = (float*)(smem + OFF_B1);
    float* b2s   = (float*)(smem + OFF_B2);
    float* b3s   = (float*)(smem + OFF_B3);

    const int tid = threadIdx.x;
    const int wv  = tid >> 5;
    const int ln  = tid & 31;
    const int nlo = ln & 15;
    const int hi  = ln >> 4;

    char*  h1wb = smem + OFF_H1 + wv * 32 * ROWB;
    char*  h2wb = smem + OFF_H2 + wv * 32 * ROWB;
    float* orw  = (float*)(smem + OFF_OROW) + wv * 32 * 20;

    const int s = blockIdx.x * NTHR + tid;        // sample owned by this lane

    const float phi0 = z[(size_t)s * LSEQ];
    float phi_prev   = phi0;
    float ldj        = 0.f;
    out_phi[(size_t)s * LSEQ] = phi0;

    for (int l = 0; l < NLAY; ++l) {
        // ---------- cooperative weight staging (transposed, f16) ----------
        __syncthreads();
        {
            const float* W2l = W2 + (size_t)l * HD * HD;
            for (int i = tid; i < HD * HD; i += NTHR) {
                int k = i >> 6, n = i & 63;                 // W2l[i] = W2[l][k][n]
                *(_Float16*)(wt2b + n * ROWB + k * 2) = (_Float16)W2l[i];
            }
            const float* W3l = W3 + (size_t)l * HD * 3 * MMIX;
            for (int i = tid; i < 32 * HD; i += NTHR) {
                int k = i >> 5, n = i & 31;
                _Float16 v = (n < 3 * MMIX) ? (_Float16)W3l[k * (3 * MMIX) + n]
                                            : (_Float16)0.f;
                *(_Float16*)(wt3b + n * ROWB + k * 2) = v;
            }
            W1s[tid] = W1[(size_t)l * 2 * HD + tid];        // NTHR == 128 == 2*HD
            if (tid < HD) {
                b1s[tid] = b1[(size_t)l * HD + tid];
                b2s[tid] = b2[(size_t)l * HD + tid];
            }
            if (tid < 32)
                b3s[tid] = (tid < 3 * MMIX) ? b3[(size_t)l * 3 * MMIX + tid] : 0.f;
        }
        __syncthreads();

        // ---------- per-lane Moebius transform ----------
        const float z_x = z[(size_t)s * LSEQ + l + 1];
        const float SU  = phi_prev - phi0;
        float V         = mod2pi(z_x - phi_prev);
        const float cs  = __cosf(SU);
        const float sn  = __sinf(SU);
        {
            float kc  = fmaxf(0.5f * __cosf(0.5f * SU), 1e-6f);    // 2*BETA = 0.5
            float rho = __expf(-2.f * __frsqrt_rn(kc));            // exp(-2/sqrt(kc))
            float xi  = mod2pi(-0.5f * SU);
            float wr  = rho * __cosf(xi), wi = rho * __sinf(xi);
            float zr  = __cosf(V), zi = __sinf(V);
            float nr  = zr - wr, ni = zi - wi;
            float dr  = 1.f - (wr * zr + wi * zi);
            float di  = -(wr * zi - wi * zr);
            float Vp  = mod2pi(fast_atan2(ni * dr - nr * di, nr * dr + ni * di));
            ldj += __logf((1.f - rho * rho) *
                          __builtin_amdgcn_rcpf(dr * dr + di * di));
            V = mod2pi(Vp - F_PI);
        }

        // ---------- layer 1 (K=2, VALU) -> h1 rows (f16 in LDS) ----------
        {
            char* myrow = h1wb + ln * ROWB;
#pragma unroll
            for (int k = 0; k < HD; k += 2) {
                float v0 = fast_tanh(fmaf(cs, W1s[k],     fmaf(sn, W1s[HD + k],     b1s[k])));
                float v1 = fast_tanh(fmaf(cs, W1s[k + 1], fmaf(sn, W1s[HD + k + 1], b1s[k + 1])));
                union { _Float16 h[2]; unsigned u; } p;
                p.h[0] = (_Float16)v0;
                p.h[1] = (_Float16)v1;
                *(unsigned*)(myrow + k * 2) = p.u;
            }
        }
        asm volatile("s_wait_dscnt 0" ::: "memory");

        // ---------- layer 2: h1[32x64] @ W2[64x64] via WMMA ----------
        {
            v16h A[2][2];
#pragma unroll
            for (int tm = 0; tm < 2; ++tm)
#pragma unroll
                for (int kc = 0; kc < 2; ++kc)
                    A[tm][kc] = loadA(h1wb, 16 * tm + nlo, kc, hi);

#pragma unroll
            for (int tn = 0; tn < 4; ++tn) {
                v16h B0 = loadB(wt2b, 16 * tn + nlo, 0, hi);
                v16h B1 = loadB(wt2b, 16 * tn + nlo, 1, hi);
                float bias = b2s[16 * tn + nlo];
#pragma unroll
                for (int tm = 0; tm < 2; ++tm) {
                    v8f c = {};
                    c = wmma_f16(A[tm][0], B0, c);
                    c = wmma_f16(A[tm][1], B1, c);
#pragma unroll
                    for (int r = 0; r < 8; ++r) {
                        float v = fast_tanh(c[r] + bias);
                        *(_Float16*)(h2wb + (16 * tm + r + 8 * hi) * ROWB +
                                     (16 * tn + nlo) * 2) = (_Float16)v;
                    }
                }
            }
        }
        asm volatile("s_wait_dscnt 0" ::: "memory");

        // ---------- layer 3: h2[32x64] @ W3[64x18pad32] via WMMA ----------
        {
            v16h A[2][2];
#pragma unroll
            for (int tm = 0; tm < 2; ++tm)
#pragma unroll
                for (int kc = 0; kc < 2; ++kc)
                    A[tm][kc] = loadA(h2wb, 16 * tm + nlo, kc, hi);

#pragma unroll
            for (int tn = 0; tn < 2; ++tn) {
                v16h B0 = loadB(wt3b, 16 * tn + nlo, 0, hi);
                v16h B1 = loadB(wt3b, 16 * tn + nlo, 1, hi);
                int col = 16 * tn + nlo;
#pragma unroll
                for (int tm = 0; tm < 2; ++tm) {
                    v8f c = {};
                    c = wmma_f16(A[tm][0], B0, c);
                    c = wmma_f16(A[tm][1], B1, c);
                    if (col < 3 * MMIX) {
#pragma unroll
                        for (int r = 0; r < 8; ++r)
                            orw[(16 * tm + r + 8 * hi) * 20 + col] = c[r];
                    }
                }
            }
        }
        asm volatile("s_wait_dscnt 0" ::: "memory");

        // ---------- per-lane mixture-of-sigmoids CDF ----------
        {
            const float* myo = orw + ln * 20;
            float o[18];
#pragma unroll
            for (int j = 0; j < 18; ++j) o[j] = myo[j] + b3s[j];

            float wmax = o[12];
#pragma unroll
            for (int j = 1; j < MMIX; ++j) wmax = fmaxf(wmax, o[12 + j]);
            float e[MMIX], esum = 0.f;
#pragma unroll
            for (int j = 0; j < MMIX; ++j) {
                e[j] = __expf(o[12 + j] - wmax);
                esum += e[j];
            }
            float rinv = __builtin_amdgcn_rcpf(esum);

            const float u = V * F_INV2PI;
            float Ux = 0.f, Hp = 0.f;
#pragma unroll
            for (int j = 0; j < MMIX; ++j) {
                float alpha = softplus_f(o[j]) + 1e-3f;
                float bb    = o[MMIX + j];
                float s0    = sigm2(-alpha * bb);
                float s1    = sigm2(alpha * (1.f - bb));
                float rd    = __builtin_amdgcn_rcpf(s1 - s0 + 1e-12f);
                float st    = sigm2(alpha * (u - bb));
                float rw    = 0.01f + (1.f - MMIX * 0.01f) * e[j] * rinv;
                Ux += rw * (st - s0) * rd;
                Hp += rw * 2.f * alpha * st * (1.f - st) * rd;
            }
            ldj += __logf(Hp);
            float phi_new = mod2pi(F_2PI * Ux + phi_prev);
            out_phi[(size_t)s * LSEQ + l + 1] = phi_new;
            phi_prev = phi_new;
        }
    }
    out_ldj[s] = ldj;
}

extern "C" void kernel_launch(void* const* d_in, const int* in_sizes, int n_in,
                              void* d_out, int out_size, void* d_ws, size_t ws_size,
                              hipStream_t stream) {
    (void)in_sizes; (void)n_in; (void)out_size; (void)d_ws; (void)ws_size;
    const float* z  = (const float*)d_in[0];
    const float* W1 = (const float*)d_in[1];
    const float* b1 = (const float*)d_in[2];
    const float* W2 = (const float*)d_in[3];
    const float* b2 = (const float*)d_in[4];
    const float* W3 = (const float*)d_in[5];
    const float* b3 = (const float*)d_in[6];
    float* out_phi = (float*)d_out;
    float* out_ldj = out_phi + (size_t)BTOT * LSEQ;

    dim3 grid(BTOT / NTHR);   // 512 blocks of 128 threads (4 waves)
    dim3 block(NTHR);
    flow_kernel<<<grid, block, SMEM_SZ, stream>>>(z, W1, b1, W2, b2, W3, b3,
                                                  out_phi, out_ldj);
}
